// HGRU_83829171683902
// MI455X (gfx1250) — matrix operand order repaired and checked by reading, not compiled, not run on hardware
//
#include <hip/hip_runtime.h>

// ---------------------------------------------------------------------------
// Persistent-kernel GRU for MI455X (gfx1250, wave32, WMMA).
//
// Grid: 16 workgroups (one 16-row batch tile each), 256 threads = 8 waves.
// All weights staged once into LDS as bf16 in WMMA B-fragment layout
// (285 KB weights + 30 KB activations + 2.6 KB biases = 325,248 B <= 320 KB
// WGP LDS). Biases staged via global_load_async_to_lds_b32 (ASYNCcnt path).
// Per time step (4 barrier-separated phases):
//   A: r,z = sigmoid([x|h] @ {r_w,z_w} + b)      (8 N-tiles, 1 per wave)
//   B: a1 = relu([x|r*h] @ h_w0 + b0)            (9 N-tiles)
//   C: a2 = relu(a1_pad160 @ h_w1_pad + b1)      (9 N-tiles)
//   D: h' = (1-z)h + z*tanh(a2_pad160 @ h_w2_pad + b2); store outputs
// z is register-resident between A and D (same wave owns same N-tile).
// x_{t+1} prefetched to registers during compute; bf16 convert at step start.
// ---------------------------------------------------------------------------

typedef __attribute__((ext_vector_type(16))) __bf16 v16bf;
typedef __attribute__((ext_vector_type(8)))  __bf16 v8bf;
typedef __attribute__((ext_vector_type(8)))  float  v8f;

namespace {
constexpr int Bb = 256, Ss = 1024, Ii = 128, Hh = 128, Mh = 144;

// LDS byte offsets. Weight blocks: 32x16 bf16 fragments = 1024 B each,
// indexed (ntile * KT + ktile).
constexpr int OFF_WR  = 0;                       // r_w  : KT=8, NT=8 -> 64 KB
constexpr int OFF_WZ  = OFF_WR  + 8 * 8 * 1024;  // z_w  : KT=8, NT=8 -> 64 KB
constexpr int OFF_WH0 = OFF_WZ  + 8 * 8 * 1024;  // h_w0 : KT=8, NT=9 -> 72 KB
constexpr int OFF_WH1 = OFF_WH0 + 8 * 9 * 1024;  // h_w1 : KT=5(pad160), NT=9
constexpr int OFF_WH2 = OFF_WH1 + 5 * 9 * 1024;  // h_w2 : KT=5(pad160), NT=8
constexpr int OFF_AX  = OFF_WH2 + 5 * 8 * 1024;  // x_t   bf16 [16][128]
constexpr int OFF_AH  = OFF_AX  + 16 * 128 * 2;  // h     bf16 [16][128]
constexpr int OFF_ARH = OFF_AH  + 16 * 128 * 2;  // r*h   bf16 [16][128]
constexpr int OFF_HF  = OFF_ARH + 16 * 128 * 2;  // h     f32  [16][128]
constexpr int OFF_A1  = OFF_HF  + 16 * 128 * 4;  // act1  bf16 [16][160] (pad 0)
constexpr int OFF_A2  = OFF_A1  + 16 * 160 * 2;  // act2  bf16 [16][160] (pad 0)
constexpr int OFF_BIAS = OFF_A2 + 16 * 160 * 2;  // 672 f32: rb,zb,b0,b1,b2
constexpr int SMEM_BYTES = OFF_BIAS + 672 * 4;   // 325,248 B <= 327,680 B

// f32 element offsets inside the bias block
constexpr int BOFF_RB = 0, BOFF_ZB = 128, BOFF_B0 = 256, BOFF_B1 = 400,
              BOFF_B2 = 544;
}

__device__ __forceinline__ unsigned short f2bf(float f) {
  union { float f; unsigned u; } v; v.f = f;
  unsigned r = v.u + 0x7FFFu + ((v.u >> 16) & 1u);   // round-to-nearest-even
  return (unsigned short)(r >> 16);
}
__device__ __forceinline__ __bf16 f2bf16(float f) {
  unsigned short b = f2bf(f);
  return __builtin_bit_cast(__bf16, b);
}
__device__ __forceinline__ unsigned pack2(float a, float b) {
  return (unsigned)f2bf(a) | ((unsigned)f2bf(b) << 16);
}

// Async global->LDS dword copy (GVS mode: SGPR base + per-lane VGPR offset).
// LDS destination address taken from the vdst VGPR; tracked by ASYNCcnt.
__device__ __forceinline__ void async_g2l_b32(const float* gbase, unsigned idx,
                                              unsigned lds_byte_off) {
  unsigned voff = idx * 4u;
  unsigned dst  = lds_byte_off + voff;   // LDS byte address
  asm volatile("global_load_async_to_lds_b32 %0, %1, %2"
               :: "v"(dst), "v"(voff), "s"(gbase)
               : "memory");
}

// A-fragment: 16x32 bf16 tile from a row-major [16][ld] bf16 LDS region.
// Lane layout per ISA: lanes 0-15 = rows, halves g=lane>>4 pick K subgroups.
__device__ __forceinline__ v16bf load_afrag(const __bf16* base, int ld, int kt,
                                            int lane) {
  int row = lane & 15, g = lane >> 4;
  int o = row * ld + kt * 32 + g * 8;
  v8bf lo = *(const v8bf*)(base + o);        // K = kt*32 + g*8      .. +7
  v8bf hi = *(const v8bf*)(base + o + 16);   // K = kt*32 + 16 + g*8 .. +7
  return __builtin_shufflevector(lo, hi, 0, 1, 2, 3, 4, 5, 6, 7,
                                         8, 9, 10, 11, 12, 13, 14, 15);
}

// B-fragment: pre-swizzled 32x16 block; lane's 16 bf16 are contiguous (32 B).
__device__ __forceinline__ v16bf load_bfrag(const __bf16* wbase, int KT, int kt,
                                            int nt, int lane) {
  const __bf16* p = wbase + ((nt * KT + kt) << 9) + lane * 16;
  v8bf lo = *(const v8bf*)(p);
  v8bf hi = *(const v8bf*)(p + 8);
  return __builtin_shufflevector(lo, hi, 0, 1, 2, 3, 4, 5, 6, 7,
                                         8, 9, 10, 11, 12, 13, 14, 15);
}

__device__ __forceinline__ v8f wmma_bf16(v16bf a, v16bf b, v8f c) {
  return __builtin_amdgcn_wmma_f32_16x16x32_bf16(
      false, a, false, b, (short)0, c, false, false);
}

// Stage a row-major f32 weight [Kdim][Ndim] into bf16 B-fragment blocks,
// zero-padding K up to KT*32 (for the MH=144 -> 160 padded layers).
__device__ void stage_weight(const float* __restrict__ W, __bf16* dst,
                             int Kdim, int Ndim, int KT, int NT, int tid) {
  int nslots = KT * NT * 32;
  for (int idx = tid; idx < nslots; idx += 256) {
    int blk = idx >> 5, ln = idx & 31;
    int nt = blk / KT, kt = blk - nt * KT;
    int col = nt * 16 + (ln & 15);
    int kbase = kt * 32 + ((ln >> 4) << 3);
    __bf16* o = dst + (blk << 9) + ln * 16;
#pragma unroll
    for (int j = 0; j < 16; ++j) {
      int k = kbase + (j & 7) + ((j >> 3) << 4);
      float v = (k < Kdim) ? W[k * Ndim + col] : 0.0f;
      o[j] = f2bf16(v);
    }
  }
}

__global__ __launch_bounds__(256) void HGRU_83829171683902_kernel(
    const float* __restrict__ inputs,
    const float* __restrict__ r_w, const float* __restrict__ r_b,
    const float* __restrict__ z_w, const float* __restrict__ z_b,
    const float* __restrict__ h_w0, const float* __restrict__ h_b0,
    const float* __restrict__ h_w1, const float* __restrict__ h_b1,
    const float* __restrict__ h_w2, const float* __restrict__ h_b2,
    float* __restrict__ out, float* __restrict__ hid) {
  extern __shared__ __align__(16) char smem[];
  __bf16* wR  = (__bf16*)(smem + OFF_WR);
  __bf16* wZ  = (__bf16*)(smem + OFF_WZ);
  __bf16* wH0 = (__bf16*)(smem + OFF_WH0);
  __bf16* wH1 = (__bf16*)(smem + OFF_WH1);
  __bf16* wH2 = (__bf16*)(smem + OFF_WH2);
  __bf16* aX  = (__bf16*)(smem + OFF_AX);
  __bf16* aH  = (__bf16*)(smem + OFF_AH);
  __bf16* aRH = (__bf16*)(smem + OFF_ARH);
  float*  hF  = (float*)(smem + OFF_HF);
  __bf16* a1  = (__bf16*)(smem + OFF_A1);
  __bf16* a2  = (__bf16*)(smem + OFF_A2);
  const float* bias = (const float*)(smem + OFF_BIAS);

  const int tid  = threadIdx.x;
  const int lane = tid & 31;
  const int wave = tid >> 5;
  const int g    = lane >> 4;
  const int m0   = blockIdx.x * 16;          // batch-tile origin

  // ---- biases -> LDS via async global->LDS (ASYNCcnt) ----------------------
  for (unsigned i = tid; i < 128; i += 256)
    async_g2l_b32(r_b,  i, OFF_BIAS + BOFF_RB * 4);
  for (unsigned i = tid; i < 128; i += 256)
    async_g2l_b32(z_b,  i, OFF_BIAS + BOFF_ZB * 4);
  for (unsigned i = tid; i < 144; i += 256)
    async_g2l_b32(h_b0, i, OFF_BIAS + BOFF_B0 * 4);
  for (unsigned i = tid; i < 144; i += 256)
    async_g2l_b32(h_b1, i, OFF_BIAS + BOFF_B1 * 4);
  for (unsigned i = tid; i < 128; i += 256)
    async_g2l_b32(h_b2, i, OFF_BIAS + BOFF_B2 * 4);

  // ---- one-time staging: weights -> LDS bf16 fragments; state = 0 ----------
  stage_weight(r_w,  wR,  256, 128, 8, 8, tid);
  stage_weight(z_w,  wZ,  256, 128, 8, 8, tid);
  stage_weight(h_w0, wH0, 256, 144, 8, 9, tid);
  stage_weight(h_w1, wH1, 144, 144, 5, 9, tid);   // K padded 144 -> 160
  stage_weight(h_w2, wH2, 144, 128, 5, 8, tid);   // K padded 144 -> 160
  for (int i = tid; i < 16 * 128; i += 256) {
    hF[i] = 0.0f; aH[i] = f2bf16(0.0f); aRH[i] = f2bf16(0.0f);
  }
  for (int i = tid; i < 16 * 160; i += 256) {     // keeps pad cols 144..159 = 0
    a1[i] = f2bf16(0.0f); a2[i] = f2bf16(0.0f);
  }

  // ---- x prefetch plumbing: each thread owns 8 contiguous x elements -------
  const int xrow = tid >> 4;                      // 0..15
  const int xcol = (tid & 15) * 8;                // 0..120
  const float* xbase =
      inputs + (size_t)(m0 + xrow) * Ss * Ii + xcol;
  float4 xp0 = *(const float4*)(xbase);
  float4 xp1 = *(const float4*)(xbase + 4);

  asm volatile("s_wait_asynccnt 0x0" ::: "memory");  // biases landed in LDS
  __syncthreads();

  float zreg[8];                                  // z, wave-resident A -> D

  for (int t = 0; t < Ss; ++t) {
    // -- commit x_t to LDS as bf16 A-operand; kick off x_{t+1} prefetch ------
    uint4 xb;
    xb.x = pack2(xp0.x, xp0.y); xb.y = pack2(xp0.z, xp0.w);
    xb.z = pack2(xp1.x, xp1.y); xb.w = pack2(xp1.z, xp1.w);
    *(uint4*)(aX + xrow * 128 + xcol) = xb;
    if (t + 1 < Ss) {
      const float* nx = xbase + (size_t)(t + 1) * Ii;
      xp0 = *(const float4*)(nx);
      xp1 = *(const float4*)(nx + 4);
    }
    if (t + 2 < Ss)                                // global_prefetch_b8
      __builtin_prefetch(xbase + (size_t)(t + 2) * Ii, 0, 3);
    __syncthreads();

    // -- Phase A: gates r, z --------------------------------------------------
    {
      const int nt = wave;                         // 8 waves cover H=128
      v8f accR = {0, 0, 0, 0, 0, 0, 0, 0};
      v8f accZ = {0, 0, 0, 0, 0, 0, 0, 0};
#pragma unroll
      for (int kt = 0; kt < 8; ++kt) {
        v16bf a = (kt < 4) ? load_afrag(aX, 128, kt, lane)
                           : load_afrag(aH, 128, kt - 4, lane);
        accR = wmma_bf16(a, load_bfrag(wR, 8, kt, nt, lane), accR);
        accZ = wmma_bf16(a, load_bfrag(wZ, 8, kt, nt, lane), accZ);
      }
      const int n = nt * 16 + (lane & 15);
      const float rb = bias[BOFF_RB + n], zb = bias[BOFF_ZB + n];
#pragma unroll
      for (int v = 0; v < 8; ++v) {
        const int m = v + 8 * g;
        float r = 1.0f / (1.0f + __expf(-(accR[v] + rb)));
        zreg[v] = 1.0f / (1.0f + __expf(-(accZ[v] + zb)));
        aRH[m * 128 + n] = f2bf16(r * hF[m * 128 + n]);
      }
    }
    __syncthreads();

    // -- Phase B: a1 = relu([x | r*h] @ h_w0 + b0) ---------------------------
    for (int nt = wave; nt < 9; nt += 8) {
      v8f acc = {0, 0, 0, 0, 0, 0, 0, 0};
#pragma unroll
      for (int kt = 0; kt < 8; ++kt) {
        v16bf a = (kt < 4) ? load_afrag(aX, 128, kt, lane)
                           : load_afrag(aRH, 128, kt - 4, lane);
        acc = wmma_bf16(a, load_bfrag(wH0, 8, kt, nt, lane), acc);
      }
      const int n = nt * 16 + (lane & 15);
      const float b0 = bias[BOFF_B0 + n];
#pragma unroll
      for (int v = 0; v < 8; ++v)
        a1[(v + 8 * g) * 160 + n] = f2bf16(fmaxf(acc[v] + b0, 0.0f));
    }
    __syncthreads();

    // -- Phase C: a2 = relu(a1 @ h_w1 + b1), K padded to 160 -----------------
    for (int nt = wave; nt < 9; nt += 8) {
      v8f acc = {0, 0, 0, 0, 0, 0, 0, 0};
#pragma unroll
      for (int kt = 0; kt < 5; ++kt)
        acc = wmma_bf16(load_afrag(a1, 160, kt, lane),
                        load_bfrag(wH1, 5, kt, nt, lane), acc);
      const int n = nt * 16 + (lane & 15);
      const float b1 = bias[BOFF_B1 + n];
#pragma unroll
      for (int v = 0; v < 8; ++v)
        a2[(v + 8 * g) * 160 + n] = f2bf16(fmaxf(acc[v] + b1, 0.0f));
    }
    __syncthreads();

    // -- Phase D: h_tilde, state update, output store ------------------------
    {
      const int nt = wave;
      v8f acc = {0, 0, 0, 0, 0, 0, 0, 0};
#pragma unroll
      for (int kt = 0; kt < 5; ++kt)
        acc = wmma_bf16(load_afrag(a2, 160, kt, lane),
                        load_bfrag(wH2, 5, kt, nt, lane), acc);
      const int n = nt * 16 + (lane & 15);
      const float b2 = bias[BOFF_B2 + n];
#pragma unroll
      for (int v = 0; v < 8; ++v) {
        const int m = v + 8 * g;
        float ht = tanhf(acc[v] + b2);
        float z  = zreg[v];
        float hn = (1.0f - z) * hF[m * 128 + n] + z * ht;
        hF[m * 128 + n] = hn;
        aH[m * 128 + n] = f2bf16(hn);
        out[((size_t)(m0 + m) * Ss + t) * Hh + n] = hn;
        if (t == Ss - 1) hid[(m0 + m) * Hh + n] = hn;
      }
    }
    __syncthreads();
  }
}

extern "C" void kernel_launch(void* const* d_in, const int* in_sizes, int n_in,
                              void* d_out, int out_size, void* d_ws,
                              size_t ws_size, hipStream_t stream) {
  const float* inputs = (const float*)d_in[0];
  const float* r_w  = (const float*)d_in[1];
  const float* r_b  = (const float*)d_in[2];
  const float* z_w  = (const float*)d_in[3];
  const float* z_b  = (const float*)d_in[4];
  const float* h_w0 = (const float*)d_in[5];
  const float* h_b0 = (const float*)d_in[6];
  const float* h_w1 = (const float*)d_in[7];
  const float* h_b1 = (const float*)d_in[8];
  const float* h_w2 = (const float*)d_in[9];
  const float* h_b2 = (const float*)d_in[10];

  float* out = (float*)d_out;                       // [B, S, H]
  float* hid = out + (size_t)Bb * Ss * Hh;          // [B, H] (tuple 2nd output)

  (void)hipFuncSetAttribute(
      reinterpret_cast<const void*>(HGRU_83829171683902_kernel),
      hipFuncAttributeMaxDynamicSharedMemorySize, SMEM_BYTES);

  HGRU_83829171683902_kernel<<<dim3(Bb / 16), dim3(256), SMEM_BYTES, stream>>>(
      inputs, r_w, r_b, z_w, z_b, h_w0, h_b0, h_w1, h_b1, h_w2, h_b2, out, hid);
}
